// QuantMatMul_3324304687440
// MI455X (gfx1250) — matrix operand order repaired and checked by reading, not compiled, not run on hardware
//
#include <hip/hip_runtime.h>

typedef __attribute__((ext_vector_type(16))) _Float16 v16h;
typedef __attribute__((ext_vector_type(8)))  _Float16 v8h;
typedef __attribute__((ext_vector_type(8)))  float    v8f;

#define QMAXF 255.0f
#define BM 128
#define BN 128
#define KD 64
#define LDA 72   // padded f16 stride: 144B rows -> 16B-aligned fragments, spread banks

// ---- order-preserving float <-> u32 key (for atomic min/max on floats) ----
__device__ __forceinline__ unsigned f2key(float f) {
  unsigned u = __float_as_uint(f);
  return (u & 0x80000000u) ? ~u : (u | 0x80000000u);
}
__device__ __forceinline__ float key2f(unsigned k) {
  return (k & 0x80000000u) ? __uint_as_float(k & 0x7fffffffu)
                           : __uint_as_float(~k);
}

__global__ void init_minmax(unsigned* ws) {
  if (threadIdx.x < 4) ws[threadIdx.x] = 0x80000000u;  // key(0.0f): min<=0, max>=0
}

__global__ __launch_bounds__(256)
void minmax_kernel(const float* __restrict__ x, int n,
                   unsigned* __restrict__ mn, unsigned* __restrict__ mx) {
  __shared__ unsigned smn[256];
  __shared__ unsigned smx[256];
  float lo = 0.0f, hi = 0.0f;
  for (int i = blockIdx.x * blockDim.x + threadIdx.x; i < n;
       i += gridDim.x * blockDim.x) {
    float v = x[i];
    lo = fminf(lo, v);
    hi = fmaxf(hi, v);
  }
  smn[threadIdx.x] = f2key(lo);
  smx[threadIdx.x] = f2key(hi);
  __syncthreads();
  for (int s = 128; s > 0; s >>= 1) {
    if ((int)threadIdx.x < s) {
      unsigned a = smn[threadIdx.x], b = smn[threadIdx.x + s];
      smn[threadIdx.x] = a < b ? a : b;
      unsigned c = smx[threadIdx.x], d = smx[threadIdx.x + s];
      smx[threadIdx.x] = c > d ? c : d;
    }
    __syncthreads();
  }
  if (threadIdx.x == 0) {
    atomicMin(mn, smn[0]);
    atomicMax(mx, smx[0]);
  }
}

// One block = 128x128 tile of one (b,h) batch. 8 waves, each wave owns 16 rows
// and sweeps eight 16x16 column sub-tiles with v_wmma_f32_16x16x32_f16.
__global__ __launch_bounds__(256)
void qmm_kernel(const float* __restrict__ A, const float* __restrict__ B,
                float* __restrict__ C, const unsigned* __restrict__ mm) {
  __shared__ _Float16 As[BM * LDA];  // centered-quant A, row-major over K
  __shared__ _Float16 Bs[BN * LDA];  // centered-quant B, N-major over K (transposed)

  const int tid = threadIdx.x;
  const int bz  = blockIdx.z;
  const int m0  = blockIdx.y * BM;
  const int n0  = blockIdx.x * BN;

  // Per-tensor quant params (uniform scalars; compiler keeps them in SGPRs).
  const float amin = key2f(mm[0]), amax = key2f(mm[1]);
  const float bmin = key2f(mm[2]), bmax = key2f(mm[3]);
  const float da  = fmaxf((amax - amin) * (1.0f / QMAXF), 1e-8f);
  const float db  = fmaxf((bmax - bmin) * (1.0f / QMAXF), 1e-8f);
  const float zpa = rintf(-amin / da);   // v_rndne == round-half-even == jnp.round
  const float zpb = rintf(-bmin / db);

  const float* Ab = A + (size_t)bz * 2048 * 64;
  const float* Bb = B + (size_t)bz * 64 * 2048;
  float*       Cb = C + (size_t)bz * 2048 * 2048;

  // Stage A tile: fake-quant, store centered integer value (exact in f16).
  for (int i = tid; i < BM * KD; i += 256) {
    int r = i >> 6, k = i & 63;
    float x = Ab[(size_t)(m0 + r) * 64 + k];
    float q = fminf(fmaxf(rintf(x / da) + zpa, 0.0f), QMAXF) - zpa;
    As[r * LDA + k] = (_Float16)q;
  }
  // Stage B tile transposed: coalesced global reads along N, write Bs[n][k].
  for (int i = tid; i < KD * BN; i += 256) {
    int k = i >> 7, n = i & 127;
    float x = Bb[(size_t)k * 2048 + n0 + n];
    float q = fminf(fmaxf(rintf(x / db) + zpb, 0.0f), QMAXF) - zpb;
    Bs[n * LDA + k] = (_Float16)q;
  }
  __syncthreads();

  const int wave   = tid >> 5;
  const int lane   = tid & 31;
  const int half16 = lane >> 4;  // 0: lanes 0-15, 1: lanes 16-31
  const int mr     = lane & 15;

  // A fragment (CDNA5 16-bit A 16x32 layout): lane group selects K-chunk of 8,
  // halves 0-7 = K kb..kb+7, halves 8-15 = K kb+16..kb+23.
  const _Float16* Arow = &As[(wave * 16 + mr) * LDA];
  v16h a0, a1;
  {
    const int kb = half16 * 8;
    *((v8h*)&a0 + 0) = *(const v8h*)&Arow[kb];        // K kb .. kb+7
    *((v8h*)&a0 + 1) = *(const v8h*)&Arow[kb + 16];   // K kb+16 .. kb+23
    *((v8h*)&a1 + 0) = *(const v8h*)&Arow[kb + 32];   // second k-step: K 32..63
    *((v8h*)&a1 + 1) = *(const v8h*)&Arow[kb + 48];
  }

  const float scale = da * db;  // fold dequant scales back in at the end

  for (int j = 0; j < 8; ++j) {
    // B fragment (16-bit B 32x16): lane = column N, same K order as A:
    // halves 0-7 = K kb..kb+7, halves 8-15 = K kb+16..kb+23.
    const _Float16* Bcol = &Bs[(j * 16 + mr) * LDA];
    const int kb = half16 * 8;
    v16h b0, b1;
    *((v8h*)&b0 + 0) = *(const v8h*)&Bcol[kb];
    *((v8h*)&b0 + 1) = *(const v8h*)&Bcol[kb + 16];
    *((v8h*)&b1 + 0) = *(const v8h*)&Bcol[kb + 32];
    *((v8h*)&b1 + 1) = *(const v8h*)&Bcol[kb + 48];

    v8f acc = {};
    acc = __builtin_amdgcn_wmma_f32_16x16x32_f16(false, a0, false, b0,
                                                 (short)0, acc, false, false);
    acc = __builtin_amdgcn_wmma_f32_16x16x32_f16(false, a1, false, b1,
                                                 (short)0, acc, false, false);

    // C/D layout: VGPR r -> M = r + 8*half16, N = lane&15.
    const int col     = n0 + j * 16 + mr;
    const int rowbase = m0 + wave * 16 + half16 * 8;
#pragma unroll
    for (int r = 0; r < 8; ++r) {
      Cb[(size_t)(rowbase + r) * 2048 + col] = acc[r] * scale;
    }
  }
}

extern "C" void kernel_launch(void* const* d_in, const int* in_sizes, int n_in,
                              void* d_out, int out_size, void* d_ws, size_t ws_size,
                              hipStream_t stream) {
  const float* A = (const float*)d_in[0];
  const float* B = (const float*)d_in[1];
  float* out     = (float*)d_out;
  unsigned* ws   = (unsigned*)d_ws;

  const int nA = in_sizes[0];
  const int nB = in_sizes[1];
  const int batches = nA / (2048 * 64);  // = 4*16

  init_minmax<<<1, 32, 0, stream>>>(ws);
  minmax_kernel<<<1024, 256, 0, stream>>>(A, nA, ws + 0, ws + 1);
  minmax_kernel<<<1024, 256, 0, stream>>>(B, nB, ws + 2, ws + 3);

  dim3 grid(2048 / BN, 2048 / BM, batches);
  qmm_kernel<<<grid, 256, 0, stream>>>(A, B, out, ws);
}
